// InteractionModule_66795331387447
// MI455X (gfx1250) — matrix operand order repaired and checked by reading, not compiled, not run on hardware
//
#include <hip/hip_runtime.h>
#include <hip/hip_bf16.h>

#define N_NODES 20000
#define EPS 1e-6f

typedef __attribute__((ext_vector_type(16))) _Float16 v16h;
typedef __attribute__((ext_vector_type(8)))  _Float16 v8h;
typedef __attribute__((ext_vector_type(8)))  float    v8f;

__device__ __forceinline__ float silu_f(float x) { return x * (1.0f / (1.0f + __expf(-x))); }

// sum across the 16 lanes that share one tile-half (all hold the same edge rows)
__device__ __forceinline__ float redux16(float v) {
  v += __shfl_xor(v, 1, 32);
  v += __shfl_xor(v, 2, 32);
  v += __shfl_xor(v, 4, 32);
  v += __shfl_xor(v, 8, 32);
  return v;
}

// A fragment (16x32 f16) from f32 global, vectorized float4 loads
__device__ __forceinline__ v16h load_a_g32v(const float* X, int ld, int r0, int k0, int lane) {
  int row = r0 + (lane & 15);
  int kb  = k0 + ((lane >> 4) << 3);
  const float4* p = (const float4*)(X + (size_t)row * ld + kb);
  float4 q0 = p[0], q1 = p[1], q2 = p[4], q3 = p[5];
  v16h a;
  a[0]  = (_Float16)q0.x; a[1]  = (_Float16)q0.y; a[2]  = (_Float16)q0.z; a[3]  = (_Float16)q0.w;
  a[4]  = (_Float16)q1.x; a[5]  = (_Float16)q1.y; a[6]  = (_Float16)q1.z; a[7]  = (_Float16)q1.w;
  a[8]  = (_Float16)q2.x; a[9]  = (_Float16)q2.y; a[10] = (_Float16)q2.z; a[11] = (_Float16)q2.w;
  a[12] = (_Float16)q3.x; a[13] = (_Float16)q3.y; a[14] = (_Float16)q3.z; a[15] = (_Float16)q3.w;
  return a;
}

// A fragment from f16 LDS [16 x ld]: two 16-byte ds loads
__device__ __forceinline__ v16h load_a_lds(const _Float16* X, int ld, int k0, int lane) {
  int row = lane & 15;
  int kb  = k0 + ((lane >> 4) << 3);
  const _Float16* p = X + row * ld + kb;
  v8h lo = *(const v8h*)p;
  v8h hi = *(const v8h*)(p + 16);
  return __builtin_shufflevector(lo, hi, 0, 1, 2, 3, 4, 5, 6, 7, 8, 9, 10, 11, 12, 13, 14, 15);
}

// B fragment from pre-packed f16 weights: one 32-byte load
__device__ __forceinline__ v16h load_b_pk(const _Float16* P, int nTiles, int kc, int n, int lane) {
  return ((const v16h*)P)[(size_t)((kc * nTiles + n) * 32 + lane)];
}

__device__ __forceinline__ v8f wmma_f16(v16h a, v16h b, v8f c) {
  return __builtin_amdgcn_wmma_f32_16x16x32_f16(false, a, false, b, (short)0, c, false, false);
}

__global__ void k_zero(float* __restrict__ p, int n) {
  int i = blockIdx.x * blockDim.x + threadIdx.x;
  if (i < n) p[i] = 0.0f;
}

// pack f32 W[K x N] -> f16 B-fragment layout:
// dst[((kc*nT + n)*32 + lane)*16 + j] = W[kc*32 + (lane>>4)*8 + (j&7) + (j>>3)*16][n*16 + (lane&15)]
__global__ void k_pack(const float* __restrict__ src, _Float16* __restrict__ dst, int K, int N) {
  int idx = blockIdx.x * blockDim.x + threadIdx.x;
  if (idx >= K * N) return;
  int j    = idx & 15;
  int lane = (idx >> 4) & 31;
  int rest = idx >> 9;
  int nT = N >> 4;
  int kc = rest / nT;
  int n  = rest - kc * nT;
  int krow = kc * 32 + ((lane >> 4) << 3) + (j & 7) + ((j >> 3) << 4);
  int col  = n * 16 + (lane & 15);
  dst[idx] = (_Float16)src[(size_t)krow * N + col];
}

// ---------------- Kernel 1: latent MLP + LayerNorm + residual, env weights, scatter ----------------
__global__ __launch_bounds__(128) void k1_latent_env(
    const int* __restrict__ edge_index, const float* __restrict__ edge_sh,
    const float* __restrict__ lat_in, const float* __restrict__ upd,
    const _Float16* __restrict__ pW1, const float* __restrict__ b1,
    const _Float16* __restrict__ pW2, const float* __restrict__ b2,
    const _Float16* __restrict__ pW3, const float* __restrict__ b3,
    const float* __restrict__ ln_g, const float* __restrict__ ln_b,
    const _Float16* __restrict__ pWenv,
    _Float16* __restrict__ ws_lat, float* __restrict__ node_s, float* __restrict__ node_v,
    int Etot)
{
  __shared__ _Float16 Hs[4][2][16 * 128];
  int lane = threadIdx.x & 31;
  int wv   = threadIdx.x >> 5;
  int e0   = (blockIdx.x * 4 + wv) << 4;
  if (e0 >= Etot) return;
  _Float16* Ha = Hs[wv][0];
  _Float16* Hb = Hs[wv][1];
  int colw = lane & 15;
  int hh   = lane >> 4;

  __builtin_prefetch(pW1, 0, 3);
  __builtin_prefetch(pW2, 0, 3);
  __builtin_prefetch(pW3, 0, 3);

  // layer 1: [16x64] @ [64x128] -> silu -> Ha
  {
    v16h a0 = load_a_g32v(lat_in, 64, e0, 0, lane);
    v16h a1 = load_a_g32v(lat_in, 64, e0, 32, lane);
#pragma unroll
    for (int n = 0; n < 8; ++n) {
      v8f c = {};
      c = wmma_f16(a0, load_b_pk(pW1, 8, 0, n, lane), c);
      c = wmma_f16(a1, load_b_pk(pW1, 8, 1, n, lane), c);
      float bb = b1[n * 16 + colw];
#pragma unroll
      for (int r = 0; r < 8; ++r)
        Ha[(r + 8 * hh) * 128 + n * 16 + colw] = (_Float16)silu_f(c[r] + bb);
    }
  }
  __builtin_amdgcn_wave_barrier();
  // layer 2: [16x128]@[128x128] -> silu -> Hb
  {
    v16h a[4];
#pragma unroll
    for (int kc = 0; kc < 4; ++kc) a[kc] = load_a_lds(Ha, 128, kc * 32, lane);
#pragma unroll
    for (int n = 0; n < 8; ++n) {
      v8f c = {};
#pragma unroll
      for (int kc = 0; kc < 4; ++kc)
        c = wmma_f16(a[kc], load_b_pk(pW2, 8, kc, n, lane), c);
      float bb = b2[n * 16 + colw];
#pragma unroll
      for (int r = 0; r < 8; ++r)
        Hb[(r + 8 * hh) * 128 + n * 16 + colw] = (_Float16)silu_f(c[r] + bb);
    }
  }
  __builtin_amdgcn_wave_barrier();
  // layer 3: [16x128]@[128x64] + bias
  v8f Y[4];
  {
    v16h a[4];
#pragma unroll
    for (int kc = 0; kc < 4; ++kc) a[kc] = load_a_lds(Hb, 128, kc * 32, lane);
#pragma unroll
    for (int n = 0; n < 4; ++n) {
      v8f c = {};
#pragma unroll
      for (int kc = 0; kc < 4; ++kc)
        c = wmma_f16(a[kc], load_b_pk(pW3, 4, kc, n, lane), c);
      float bb = b3[n * 16 + colw];
#pragma unroll
      for (int r = 0; r < 8; ++r) Y[n][r] = c[r] + bb;
    }
  }
  // LayerNorm (per edge row over 64 cols) + residual mixing; stage lat for W_env matmul
  float u  = upd[0];
  float co = rsqrtf(u * u + 1.0f);
  float cn = u * co;
  float gl[4], bl[4];
#pragma unroll
  for (int n = 0; n < 4; ++n) { gl[n] = ln_g[n * 16 + colw]; bl[n] = ln_b[n * 16 + colw]; }
#pragma unroll
  for (int r = 0; r < 8; ++r) {
    float s = 0.f, s2 = 0.f;
#pragma unroll
    for (int n = 0; n < 4; ++n) { float x = Y[n][r]; s += x; s2 += x * x; }
    s  = redux16(s);
    s2 = redux16(s2);
    float mu = s * (1.0f / 64.0f);
    float iv = rsqrtf(s2 * (1.0f / 64.0f) - mu * mu + 1e-5f);
    int row = r + 8 * hh;
#pragma unroll
    for (int n = 0; n < 4; ++n) {
      int col = n * 16 + colw;
      float nl = (Y[n][r] - mu) * iv * gl[n] + bl[n];
      float l  = co * lat_in[(size_t)(e0 + row) * 64 + col] + cn * nl;
      _Float16 lh = (_Float16)l;
      ws_lat[(size_t)(e0 + row) * 64 + col] = lh;  // f16 latent for pass 2
      Ha[row * 64 + col] = lh;                     // stage (ld=64) for W_env
    }
  }
  __builtin_amdgcn_wave_barrier();
  // env weights: [16x64]@[64x64]
  v8f Wt[4];
  {
    v16h a0 = load_a_lds(Ha, 64, 0, lane);
    v16h a1 = load_a_lds(Ha, 64, 32, lane);
#pragma unroll
    for (int n = 0; n < 4; ++n) {
      v8f c = {};
      c = wmma_f16(a0, load_b_pk(pWenv, 4, 0, n, lane), c);
      c = wmma_f16(a1, load_b_pk(pWenv, 4, 1, n, lane), c);
      Wt[n] = c;
    }
  }
  // scatter env contributions (col parity: even = l=0 weight, odd = l=1 weight)
  int parity = colw & 1;
  int cent[8];
  float sh[8][4];
#pragma unroll
  for (int r = 0; r < 8; ++r) {
    int e = e0 + r + 8 * hh;
    cent[r] = edge_index[e];
    float4 s4 = ((const float4*)edge_sh)[e];
    sh[r][0] = s4.x; sh[r][1] = s4.y; sh[r][2] = s4.z; sh[r][3] = s4.w;
  }
#pragma unroll
  for (int n = 0; n < 4; ++n) {
#pragma unroll
    for (int r = 0; r < 8; ++r) {
      int m = (n * 16 + colw) >> 1;
      float wv_ = Wt[n][r];
      int c = cent[r];
      if (parity == 0) {
        atomicAdd(&node_s[c * 32 + m], sh[r][0] * wv_);
      } else {
        float* pv = &node_v[(c * 32 + m) * 3];
        atomicAdd(pv + 0, sh[r][1] * wv_);
        atomicAdd(pv + 1, sh[r][2] * wv_);
        atomicAdd(pv + 2, sh[r][3] * wv_);
      }
    }
  }
}

// ---------------- Kernel 2: node SO(3) norm + channel mix (env_linear) ----------------
__global__ __launch_bounds__(256) void k2_node(
    const float* __restrict__ node_s, const float* __restrict__ node_v,
    const float* __restrict__ lin_s, const float* __restrict__ lin_v,
    float* __restrict__ node_s2, float* __restrict__ node_v2)
{
  __shared__ float Sb[8][32];
  __shared__ float Vsb[8][3][32];
  int lane = threadIdx.x & 31;
  int wv   = threadIdx.x >> 5;
  int nd   = blockIdx.x * 8 + wv;
  if (nd >= N_NODES) return;
  float s  = node_s[nd * 32 + lane];
  float v0 = node_v[(nd * 32 + lane) * 3 + 0];
  float v1 = node_v[(nd * 32 + lane) * 3 + 1];
  float v2 = node_v[(nd * 32 + lane) * 3 + 2];
  float rs = s * s;
  float rv = v0 * v0 + v1 * v1 + v2 * v2;
#pragma unroll
  for (int m = 1; m < 32; m <<= 1) { rs += __shfl_xor(rs, m, 32); rv += __shfl_xor(rv, m, 32); }
  float is  = rsqrtf(rs * (1.0f / 32.0f) + EPS);
  float ivn = rsqrtf(rv * (1.0f / 96.0f) + EPS);
  Sb[wv][lane]     = s  * is;
  Vsb[wv][0][lane] = v0 * ivn;
  Vsb[wv][1][lane] = v1 * ivn;
  Vsb[wv][2][lane] = v2 * ivn;
  __builtin_amdgcn_wave_barrier();
  float os = 0.f, o0 = 0.f, o1 = 0.f, o2 = 0.f;
#pragma unroll 8
  for (int m = 0; m < 32; ++m) {
    float wS = lin_s[m * 32 + lane];
    float wV = lin_v[m * 32 + lane];
    os += Sb[wv][m] * wS;
    o0 += Vsb[wv][0][m] * wV;
    o1 += Vsb[wv][1][m] * wV;
    o2 += Vsb[wv][2][m] * wV;
  }
  node_s2[nd * 32 + lane] = os;
  node_v2[(nd * 32 + lane) * 3 + 0] = o0;
  node_v2[(nd * 32 + lane) * 3 + 1] = o1;
  node_v2[(nd * 32 + lane) * 3 + 2] = o2;
}

// ---------------- Kernel 3: tensor products + projection MLP + equivariant mix ----------------
__global__ __launch_bounds__(128) void k3_tp_out(
    const int* __restrict__ edge_index, const float* __restrict__ edge_sh,
    const _Float16* __restrict__ ws_lat, const _Float16* __restrict__ pWenv,
    const float* __restrict__ node_s2, const float* __restrict__ node_v2,
    const _Float16* __restrict__ pWp1, const float* __restrict__ bp1,
    const _Float16* __restrict__ pWp2, const float* __restrict__ bp2,
    const _Float16* __restrict__ pWp3, const float* __restrict__ bp3,
    const _Float16* __restrict__ pEq,
    float* __restrict__ out_scalar, float* __restrict__ out_eq, int Etot)
{
  __shared__ _Float16 Pb[4][16 * 128];
  __shared__ _Float16 Vb[4][3][16 * 96];
  __shared__ float    Inv[4][5][16];
  int lane = threadIdx.x & 31;
  int wv   = threadIdx.x >> 5;
  int e0   = (blockIdx.x * 4 + wv) << 4;
  if (e0 >= Etot) return;
  _Float16* Pw = Pb[wv];
  int colw = lane & 15;
  int hh   = lane >> 4;

  __builtin_prefetch(pWp1, 0, 3);
  __builtin_prefetch(pWp2, 0, 3);

  // stage f16 lat into P cols 0..63 via CDNA5 async global->LDS copy (16-byte granules)
  {
    unsigned lbase = (unsigned)(size_t)(void*)Pw;          // low 32 bits = LDS offset
    const _Float16* gsrc = ws_lat + (size_t)e0 * 64;
#pragma unroll
    for (int t = lane; t < 128; t += 32) {
      int row = t >> 3, seg = t & 7;
      unsigned laddr = lbase + (unsigned)(row * 256 + seg * 16);
      unsigned long long gaddr = (unsigned long long)(size_t)(const void*)(gsrc + row * 64 + seg * 8);
      asm volatile("global_load_async_to_lds_b128 %0, %1, off"
                   :: "v"(laddr), "v"(gaddr) : "memory");
    }
    asm volatile("s_wait_asynccnt 0" ::: "memory");
  }
  __builtin_amdgcn_wave_barrier();

  // w = lat @ W_env
  v8f Wt[4];
  {
    v16h a0 = load_a_lds(Pw, 128, 0, lane);
    v16h a1 = load_a_lds(Pw, 128, 32, lane);
#pragma unroll
    for (int n = 0; n < 4; ++n) {
      v8f c = {};
      c = wmma_f16(a0, load_b_pk(pWenv, 4, 0, n, lane), c);
      c = wmma_f16(a1, load_b_pk(pWenv, 4, 1, n, lane), c);
      Wt[n] = c;
    }
  }
  int parity = colw & 1;
  int cent[8];
  float sh[8][4];
#pragma unroll
  for (int r = 0; r < 8; ++r) {
    int e = e0 + r + 8 * hh;
    cent[r] = edge_index[e];
    float4 s4 = ((const float4*)edge_sh)[e];
    sh[r][0] = s4.x; sh[r][1] = s4.y; sh[r][2] = s4.z; sh[r][3] = s4.w;
  }
  // per-edge inverse norms for f_s / f_v
  float ifs[8], ifv[8];
#pragma unroll
  for (int r = 0; r < 8; ++r) {
    float ps = 0.f, pv = 0.f;
#pragma unroll
    for (int n = 0; n < 4; ++n) {
      float x = Wt[n][r] * Wt[n][r];
      if (parity == 0) ps += x; else pv += x;
    }
    ps = redux16(ps);
    pv = redux16(pv);
    float s0sq = sh[r][0] * sh[r][0];
    float svsq = sh[r][1] * sh[r][1] + sh[r][2] * sh[r][2] + sh[r][3] * sh[r][3];
    ifs[r] = rsqrtf(s0sq * ps * (1.0f / 32.0f) + EPS);
    ifv[r] = rsqrtf(svsq * pv * (1.0f / 96.0f) + EPS);
  }
  // raw tensor products into LDS + per-edge sum-of-squares
  float a_s1[8], a_s2[8], a_v1[8], a_v2[8], a_v3[8];
#pragma unroll
  for (int r = 0; r < 8; ++r) { a_s1[r] = a_s2[r] = a_v1[r] = a_v2[r] = a_v3[r] = 0.f; }
#pragma unroll
  for (int n = 0; n < 4; ++n) {
#pragma unroll
    for (int r = 0; r < 8; ++r) {
      int row = r + 8 * hh;
      int m = (n * 16 + colw) >> 1;
      int c = cent[r];
      float wv_ = Wt[n][r];
      float es  = node_s2[c * 32 + m];
      const float* pv = &node_v2[(c * 32 + m) * 3];
      float ev0 = pv[0], ev1 = pv[1], ev2 = pv[2];
      if (parity == 0) {
        float fs = sh[r][0] * wv_ * ifs[r];
        float t1 = fs * es;                                  // 0e x 0e
        Pw[row * 128 + 64 + m] = (_Float16)t1;
        a_s1[r] += t1 * t1;
        float u0 = fs * ev0, u1 = fs * ev1, u2 = fs * ev2;   // 0e x 1o
        Vb[wv][0][row * 96 + m] = (_Float16)u0;
        Vb[wv][1][row * 96 + m] = (_Float16)u1;
        Vb[wv][2][row * 96 + m] = (_Float16)u2;
        a_v1[r] += u0 * u0 + u1 * u1 + u2 * u2;
      } else {
        float f0 = sh[r][1] * wv_ * ifv[r];
        float f1 = sh[r][2] * wv_ * ifv[r];
        float f2 = sh[r][3] * wv_ * ifv[r];
        float t2 = (f0 * ev0 + f1 * ev1 + f2 * ev2) * 0.57735027f;  // 1o.1o -> 0e
        Pw[row * 128 + 96 + m] = (_Float16)t2;
        a_s2[r] += t2 * t2;
        float w0 = f0 * es, w1 = f1 * es, w2 = f2 * es;             // 1o x 0e
        Vb[wv][0][row * 96 + 32 + m] = (_Float16)w0;
        Vb[wv][1][row * 96 + 32 + m] = (_Float16)w1;
        Vb[wv][2][row * 96 + 32 + m] = (_Float16)w2;
        a_v2[r] += w0 * w0 + w1 * w1 + w2 * w2;
        float x0 = (f1 * ev2 - f2 * ev1) * 0.70710678f;             // cross / sqrt2
        float x1 = (f2 * ev0 - f0 * ev2) * 0.70710678f;
        float x2 = (f0 * ev1 - f1 * ev0) * 0.70710678f;
        Vb[wv][0][row * 96 + 64 + m] = (_Float16)x0;
        Vb[wv][1][row * 96 + 64 + m] = (_Float16)x1;
        Vb[wv][2][row * 96 + 64 + m] = (_Float16)x2;
        a_v3[r] += x0 * x0 + x1 * x1 + x2 * x2;
      }
    }
  }
#pragma unroll
  for (int r = 0; r < 8; ++r) {
    float r1 = redux16(a_s1[r]);
    float r2 = redux16(a_s2[r]);
    float r3 = redux16(a_v1[r]);
    float r4 = redux16(a_v2[r]);
    float r5 = redux16(a_v3[r]);
    if (colw == 0) {
      int row = r + 8 * hh;
      Inv[wv][0][row] = rsqrtf(r1 * (1.0f / 32.0f) + EPS);
      Inv[wv][1][row] = rsqrtf(r2 * (1.0f / 32.0f) + EPS);
      Inv[wv][2][row] = rsqrtf(r3 * (1.0f / 96.0f) + EPS);
      Inv[wv][3][row] = rsqrtf(r4 * (1.0f / 96.0f) + EPS);
      Inv[wv][4][row] = rsqrtf(r5 * (1.0f / 96.0f) + EPS);
    }
  }
  __builtin_amdgcn_wave_barrier();
  // normalize scalar TP features in P
  for (int t = lane; t < 16 * 64; t += 32) {
    int row = t >> 6, cc = t & 63;
    float f = Inv[wv][cc < 32 ? 0 : 1][row];
    int idx = row * 128 + 64 + cc;
    Pw[idx] = (_Float16)((float)Pw[idx] * f);
  }
  // normalize vector TP features in V
  for (int t = lane; t < 16 * 96; t += 32) {
    int row = t / 96, cc = t - row * 96;
    float f = Inv[wv][2 + cc / 32][row];
    int idx = row * 96 + cc;
#pragma unroll
    for (int i = 0; i < 3; ++i)
      Vb[wv][i][idx] = (_Float16)((float)Vb[wv][i][idx] * f);
  }
  __builtin_amdgcn_wave_barrier();

  // equivariant output: [16x96]@[96x32] per vector component
#pragma unroll
  for (int i = 0; i < 3; ++i) {
    v16h a[3];
#pragma unroll
    for (int kc = 0; kc < 3; ++kc) a[kc] = load_a_lds(Vb[wv][i], 96, kc * 32, lane);
#pragma unroll
    for (int n = 0; n < 2; ++n) {
      v8f c = {};
#pragma unroll
      for (int kc = 0; kc < 3; ++kc)
        c = wmma_f16(a[kc], load_b_pk(pEq, 2, kc, n, lane), c);
#pragma unroll
      for (int r = 0; r < 8; ++r) {
        int row = r + 8 * hh;
        out_eq[((size_t)(e0 + row) * 32 + n * 16 + colw) * 3 + i] = c[r];
      }
    }
  }
  __builtin_amdgcn_wave_barrier();

  // projection MLP: P[16x128] -> 128 -> 128 -> 64  (reuse V space as hidden buffer)
  _Float16* Hh = &Vb[wv][0][0];
  {
    v16h a[4];
#pragma unroll
    for (int kc = 0; kc < 4; ++kc) a[kc] = load_a_lds(Pw, 128, kc * 32, lane);
#pragma unroll
    for (int n = 0; n < 8; ++n) {
      v8f c = {};
#pragma unroll
      for (int kc = 0; kc < 4; ++kc)
        c = wmma_f16(a[kc], load_b_pk(pWp1, 8, kc, n, lane), c);
      float bb = bp1[n * 16 + colw];
#pragma unroll
      for (int r = 0; r < 8; ++r)
        Hh[(r + 8 * hh) * 128 + n * 16 + colw] = (_Float16)silu_f(c[r] + bb);
    }
  }
  __builtin_amdgcn_wave_barrier();
  {
    v16h a[4];
#pragma unroll
    for (int kc = 0; kc < 4; ++kc) a[kc] = load_a_lds(Hh, 128, kc * 32, lane);
#pragma unroll
    for (int n = 0; n < 8; ++n) {
      v8f c = {};
#pragma unroll
      for (int kc = 0; kc < 4; ++kc)
        c = wmma_f16(a[kc], load_b_pk(pWp2, 8, kc, n, lane), c);
      float bb = bp2[n * 16 + colw];
#pragma unroll
      for (int r = 0; r < 8; ++r)
        Hh[(r + 8 * hh) * 128 + n * 16 + colw] = (_Float16)silu_f(c[r] + bb);
    }
  }
  __builtin_amdgcn_wave_barrier();
  {
    v16h a[4];
#pragma unroll
    for (int kc = 0; kc < 4; ++kc) a[kc] = load_a_lds(Hh, 128, kc * 32, lane);
#pragma unroll
    for (int n = 0; n < 4; ++n) {
      v8f c = {};
#pragma unroll
      for (int kc = 0; kc < 4; ++kc)
        c = wmma_f16(a[kc], load_b_pk(pWp3, 4, kc, n, lane), c);
      float bb = bp3[n * 16 + colw];
#pragma unroll
      for (int r = 0; r < 8; ++r) {
        int row = r + 8 * hh;
        out_scalar[(size_t)(e0 + row) * 64 + n * 16 + colw] = c[r] + bb;
      }
    }
  }
}

extern "C" void kernel_launch(void* const* d_in, const int* in_sizes, int n_in,
                              void* d_out, int out_size, void* d_ws, size_t ws_size,
                              hipStream_t stream)
{
  const int*   edge_index    = (const int*)d_in[0];
  const float* edge_sh       = (const float*)d_in[1];
  const float* scalar_latent = (const float*)d_in[2];
  // d_in[3] = active_edges == arange(E): residual applies to every edge
  const float* upd  = (const float*)d_in[4];
  const float* W1   = (const float*)d_in[5];
  const float* b1   = (const float*)d_in[6];
  const float* W2   = (const float*)d_in[7];
  const float* b2   = (const float*)d_in[8];
  const float* W3   = (const float*)d_in[9];
  const float* b3   = (const float*)d_in[10];
  const float* lng  = (const float*)d_in[11];
  const float* lnb  = (const float*)d_in[12];
  const float* Wenv = (const float*)d_in[13];
  const float* lins = (const float*)d_in[14];
  const float* linv = (const float*)d_in[15];
  const float* Wp1  = (const float*)d_in[16];
  const float* bp1  = (const float*)d_in[17];
  const float* Wp2  = (const float*)d_in[18];
  const float* bp2  = (const float*)d_in[19];
  const float* Wp3  = (const float*)d_in[20];
  const float* bp3  = (const float*)d_in[21];
  const float* eqlw = (const float*)d_in[22];

  int E = in_sizes[2] / 64;

  // workspace layout: f32 node buffers | f16 latent | f16 packed weights
  float* ws      = (float*)d_ws;
  float* node_s  = ws;                                   // N*32
  float* node_v  = node_s + (size_t)N_NODES * 32;        // N*96 (contiguous with node_s)
  float* node_s2 = node_v + (size_t)N_NODES * 96;        // N*32
  float* node_v2 = node_s2 + (size_t)N_NODES * 32;       // N*96
  _Float16* ws_lat = (_Float16*)(node_v2 + (size_t)N_NODES * 96);  // E*64 halves
  _Float16* wpk  = ws_lat + (size_t)E * 64;
  _Float16* pW1  = wpk;                 // 64*128
  _Float16* pW2  = pW1 + 64 * 128;      // 128*128
  _Float16* pW3  = pW2 + 128 * 128;     // 128*64
  _Float16* pWenv= pW3 + 128 * 64;      // 64*64
  _Float16* pWp1 = pWenv + 64 * 64;     // 128*128
  _Float16* pWp2 = pWp1 + 128 * 128;    // 128*128
  _Float16* pWp3 = pWp2 + 128 * 128;    // 128*64
  _Float16* pEq  = pWp3 + 128 * 64;     // 96*32

  float* out_scalar = (float*)d_out;
  float* out_eq     = out_scalar + (size_t)E * 64;

  // pack weights to f16 B-fragment layout (tiny, L2-resident afterwards)
  k_pack<<<(64 * 128 + 255) / 256, 256, 0, stream>>>(W1,   pW1,  64, 128);
  k_pack<<<(128 * 128 + 255) / 256, 256, 0, stream>>>(W2,  pW2, 128, 128);
  k_pack<<<(128 * 64 + 255) / 256, 256, 0, stream>>>(W3,   pW3, 128, 64);
  k_pack<<<(64 * 64 + 255) / 256, 256, 0, stream>>>(Wenv,  pWenv, 64, 64);
  k_pack<<<(128 * 128 + 255) / 256, 256, 0, stream>>>(Wp1, pWp1, 128, 128);
  k_pack<<<(128 * 128 + 255) / 256, 256, 0, stream>>>(Wp2, pWp2, 128, 128);
  k_pack<<<(128 * 64 + 255) / 256, 256, 0, stream>>>(Wp3,  pWp3, 128, 64);
  k_pack<<<(96 * 32 + 255) / 256, 256, 0, stream>>>(eqlw,  pEq,  96, 32);

  int nacc = N_NODES * 128;
  k_zero<<<(nacc + 255) / 256, 256, 0, stream>>>(node_s, nacc);

  int nblk = (E + 63) / 64;   // 4 waves/block, 16 edges/wave
  k1_latent_env<<<nblk, 128, 0, stream>>>(edge_index, edge_sh, scalar_latent, upd,
      pW1, b1, pW2, b2, pW3, b3, lng, lnb, pWenv, ws_lat, node_s, node_v, E);
  k2_node<<<(N_NODES + 7) / 8, 256, 0, stream>>>(node_s, node_v, lins, linv, node_s2, node_v2);
  k3_tp_out<<<nblk, 128, 0, stream>>>(edge_index, edge_sh, ws_lat, pWenv, node_s2, node_v2,
      pWp1, bp1, pWp2, bp2, pWp3, bp3, pEq, out_scalar, out_eq, E);
}
